// MultiHeadedAttention_78512002171164
// MI455X (gfx1250) — matrix operand order repaired and checked by reading, not compiled, not run on hardware
//
#include <hip/hip_runtime.h>
#include <hip/hip_bf16.h>

typedef __bf16 bf16_t;
typedef __attribute__((ext_vector_type(16))) __bf16 v16bf;
typedef __attribute__((ext_vector_type(8)))  __bf16 v8bf;
typedef __attribute__((ext_vector_type(4)))  __bf16 v4bf;
typedef __attribute__((ext_vector_type(8)))  float  v8f;
typedef __attribute__((ext_vector_type(4)))  float  v4f;

union V16 { v16bf v; v8bf h[2]; };

#define DEV static __device__ __forceinline__

DEV bf16_t f2bf(float f) { return (bf16_t)f; }   // hardware v_cvt (RNE)

constexpr int B = 4, S = 2048, E = 768, H = 3, D = 256;
constexpr int EO = H * D; // 768
constexpr float SCALE = 0.022097086912079608f;   // 1/sqrt(S), per reference

// ---------------------------------------------------------------------------
// Kernel 0: one-time weight conversion.
//   which 0..2 : M[H,E,D] fp32 -> Mt[H,D,E] bf16 (transposed, B-frag friendly)
//   which 3    : Wo[E,E]  fp32 -> WoT[n][k] = bf16(Wo[k][n])
// ---------------------------------------------------------------------------
__global__ __launch_bounds__(256)
void convert_weights_kernel(const float* __restrict__ MKp, const float* __restrict__ MVp,
                            const float* __restrict__ MQp, const float* __restrict__ Wo,
                            bf16_t* __restrict__ Mt, bf16_t* __restrict__ WoT)
{
  const int which = blockIdx.z;
  const int idx = blockIdx.x * 256 + threadIdx.x;
  if (which < 3) {
    const float* src = (which == 0) ? MKp : ((which == 1) ? MVp : MQp);
    if (idx < H * D * E) {
      const int e = idx % E;
      const int d = (idx / E) % D;
      const int h = idx / (E * D);
      Mt[(size_t)which * H * D * E + idx] = f2bf(src[((size_t)h * E + e) * D + d]);
    }
  } else {
    if (idx < EO * EO) {
      const int k = idx % EO, n = idx / EO;
      WoT[idx] = f2bf(Wo[(size_t)k * EO + n]);
    }
  }
}

// ---------------------------------------------------------------------------
// Kernel 1: per-head QKV projections, bf16 WMMA, 32x256 output block per
// 256-thread workgroup.  A-tile (X fp32) staged via ping-pong LDS buffers
// (one barrier per stage); each wave does 4 WMMAs per stage with B reuse.
// Q,K -> [B,H,S,D] (Q pre-scaled by 1/sqrt(S)); V -> transposed [B,H,D,S].
// grid: (S/32, B*H, 3), block 256 (8 waves, each wave one 32-col strip).
// ---------------------------------------------------------------------------
__global__ __launch_bounds__(256)
void qkv_proj_kernel(const float* __restrict__ Xk, const float* __restrict__ Xv,
                     const float* __restrict__ Xq, const bf16_t* __restrict__ Mt,
                     bf16_t* __restrict__ Kb, bf16_t* __restrict__ Vt,
                     bf16_t* __restrict__ Qb)
{
  __shared__ __align__(16) bf16_t xs[2][32][32];
  const int which = blockIdx.z;                 // 0=K, 1=V(transposed), 2=Q
  const float* X = (which == 0) ? Xk : ((which == 1) ? Xv : Xq);
  const bf16_t* M = Mt + (size_t)which * H * D * E;
  const int bh = blockIdx.y;
  const int h  = bh % H;
  const int s0 = blockIdx.x * 32;
  const int wave = threadIdx.x >> 5, lane = threadIdx.x & 31;
  const int hf = lane >> 4, lm = lane & 15;

  // coalesced staging: each thread converts 4 consecutive elements
  const int crow = threadIdx.x >> 3;            // 32 rows
  const int ccol = (threadIdx.x & 7) * 4;       // 32 cols
  const float* xsrc = X + ((size_t)(bh / H) * S + s0 + crow) * E + ccol;

  const int d0a = wave * 32;                    // two 16-col D tiles per wave
  const int d0b = d0a + 16;
  const bf16_t* mrow_a = M + ((size_t)h * D + d0a + lm) * E;
  const bf16_t* mrow_b = M + ((size_t)h * D + d0b + lm) * E;

  v8f acc[2][2] = {};

  // stage tile 0
  {
    v4f x = *(const v4f*)(xsrc);
    v4bf c; c[0] = f2bf(x[0]); c[1] = f2bf(x[1]); c[2] = f2bf(x[2]); c[3] = f2bf(x[3]);
    *(v4bf*)(&xs[0][crow][ccol]) = c;
  }
  __syncthreads();

  constexpr int NIT = E / 32;                   // 24
  for (int ic = 0; ic < NIT; ++ic) {
    const int ec = ic * 32;
    const int cur = ic & 1, nxt = cur ^ 1;
    v4f xn = {};
    if (ic + 1 < NIT) xn = *(const v4f*)(xsrc + ec + 32);

    v16bf b0 = *(const v16bf*)(mrow_a + ec + hf * 16);
    v16bf b1 = *(const v16bf*)(mrow_b + ec + hf * 16);
    V16 a0, a1;
    a0.h[0] = *(const v8bf*)(&xs[cur][lm][hf * 16]);
    a0.h[1] = *(const v8bf*)(&xs[cur][lm][hf * 16 + 8]);
    a1.h[0] = *(const v8bf*)(&xs[cur][16 + lm][hf * 16]);
    a1.h[1] = *(const v8bf*)(&xs[cur][16 + lm][hf * 16 + 8]);

    acc[0][0] = __builtin_amdgcn_wmma_f32_16x16x32_bf16(false, a0.v, false, b0,
                                                        (short)0, acc[0][0], false, false);
    acc[0][1] = __builtin_amdgcn_wmma_f32_16x16x32_bf16(false, a0.v, false, b1,
                                                        (short)0, acc[0][1], false, false);
    acc[1][0] = __builtin_amdgcn_wmma_f32_16x16x32_bf16(false, a1.v, false, b0,
                                                        (short)0, acc[1][0], false, false);
    acc[1][1] = __builtin_amdgcn_wmma_f32_16x16x32_bf16(false, a1.v, false, b1,
                                                        (short)0, acc[1][1], false, false);

    if (ic + 1 < NIT) {                         // publish next tile (other buffer)
      v4bf c; c[0] = f2bf(xn[0]); c[1] = f2bf(xn[1]); c[2] = f2bf(xn[2]); c[3] = f2bf(xn[3]);
      *(v4bf*)(&xs[nxt][crow][ccol]) = c;
    }
    __syncthreads();
  }

  const float osc = (which == 2) ? SCALE : 1.0f;   // pre-scale Q
  if (which != 1) {                             // Q or K: row-major [S,D]
    bf16_t* O = ((which == 0) ? Kb : Qb) + (size_t)bh * S * D;
#pragma unroll
    for (int r = 0; r < 2; ++r)
#pragma unroll
      for (int j = 0; j < 8; ++j) {
        const int row = s0 + r * 16 + hf * 8 + j;
        O[(size_t)row * D + d0a + lm] = f2bf(acc[r][0][j] * osc);
        O[(size_t)row * D + d0b + lm] = f2bf(acc[r][1][j] * osc);
      }
  } else {                                      // V: transposed [D,S]
    bf16_t* O = Vt + (size_t)bh * D * S;
#pragma unroll
    for (int r = 0; r < 2; ++r)
#pragma unroll
      for (int j = 0; j < 8; ++j) {
        const int row = s0 + r * 16 + hf * 8 + j;
        O[(size_t)(d0a + lm) * S + row] = f2bf(acc[r][0][j]);
        O[(size_t)(d0b + lm) * S + row] = f2bf(acc[r][1][j]);
      }
  }
}

// ---------------------------------------------------------------------------
// Kernel 2: causal flash attention, one wave per 16-query tile.
// Q is pre-scaled by 1/sqrt(S), so scores come out of WMMA ready to mask.
// grid: (S/64, B*H), block 128 (4 waves).  K processed in blocks of 32.
// ---------------------------------------------------------------------------
__global__ __launch_bounds__(128)
void attn_kernel(const bf16_t* __restrict__ Qb, const bf16_t* __restrict__ Kb,
                 const bf16_t* __restrict__ Vt, bf16_t* __restrict__ Ob)
{
  __shared__ __align__(32) bf16_t pl[4][16][32];   // per-wave P staging
  const int bh = blockIdx.y;
  const int b = bh / H, h = bh % H;
  const int wave = threadIdx.x >> 5, lane = threadIdx.x & 31;
  const int hf = lane >> 4, lm = lane & 15;
  const int q0 = blockIdx.x * 64 + wave * 16;

  const bf16_t* qbase = Qb + (size_t)bh * S * D;
  const bf16_t* kbase = Kb + (size_t)bh * S * D;
  const bf16_t* vbase = Vt + (size_t)bh * D * S;

  // Q A-fragments for all 8 d-chunks, kept resident.
  V16 qf[8];
  {
    const bf16_t* qrow = qbase + (size_t)(q0 + lm) * D;
#pragma unroll
    for (int c = 0; c < 8; ++c) {
      qf[c].h[0] = *(const v8bf*)(qrow + c * 32 + hf * 16);
      qf[c].h[1] = *(const v8bf*)(qrow + c * 32 + hf * 16 + 8);
    }
  }

  float m[8], l[8];
  v8f o[16] = {};
#pragma unroll
  for (int j = 0; j < 8; ++j) { m[j] = -3.0e38f; l[j] = 0.0f; }

  const int nkb = (q0 + 15) / 32 + 1;

  for (int kb = 0; kb < nkb; ++kb) {
    const int k0 = kb * 32;
    v8f s0 = {}; v8f s1 = {};
#pragma unroll
    for (int c = 0; c < 8; ++c) {
      v16bf kf0 = *(const v16bf*)(kbase + (size_t)(k0 + lm) * D      + c * 32 + hf * 16);
      v16bf kf1 = *(const v16bf*)(kbase + (size_t)(k0 + 16 + lm) * D + c * 32 + hf * 16);
      s0 = __builtin_amdgcn_wmma_f32_16x16x32_bf16(false, qf[c].v, false, kf0,
                                                   (short)0, s0, false, false);
      s1 = __builtin_amdgcn_wmma_f32_16x16x32_bf16(false, qf[c].v, false, kf1,
                                                   (short)0, s1, false, false);
    }
    // causal mask + online softmax (rows live across 16 lanes)
#pragma unroll
    for (int j = 0; j < 8; ++j) {
      const int row = q0 + hf * 8 + j;
      float a0 = s0[j], a1 = s1[j];
      if (k0 + lm      > row) a0 = -3.0e38f;
      if (k0 + 16 + lm > row) a1 = -3.0e38f;
      float mx = fmaxf(a0, a1);
#pragma unroll
      for (int off = 1; off < 16; off <<= 1) mx = fmaxf(mx, __shfl_xor(mx, off, 32));
      const float mnew = fmaxf(m[j], mx);
      const float corr = __expf(m[j] - mnew);
      m[j] = mnew;
      const float p0 = __expf(a0 - mnew);
      const float p1 = __expf(a1 - mnew);
      float rs = p0 + p1;
#pragma unroll
      for (int off = 1; off < 16; off <<= 1) rs += __shfl_xor(rs, off, 32);
      l[j] = l[j] * corr + rs;
#pragma unroll
      for (int dt = 0; dt < 16; ++dt) o[dt][j] *= corr;
      pl[wave][hf * 8 + j][lm]      = f2bf(p0);
      pl[wave][hf * 8 + j][16 + lm] = f2bf(p1);
    }
    // read back P in A-fragment layout (intra-wave DS ops are in-order)
    V16 pf;
    pf.h[0] = *(const v8bf*)(&pl[wave][lm][hf * 16]);
    pf.h[1] = *(const v8bf*)(&pl[wave][lm][hf * 16 + 8]);
    // O += P @ V  (V transposed: contiguous 32B per lane)
#pragma unroll
    for (int dt = 0; dt < 16; ++dt) {
      v16bf vf = *(const v16bf*)(vbase + (size_t)(dt * 16 + lm) * S + k0 + hf * 16);
      o[dt] = __builtin_amdgcn_wmma_f32_16x16x32_bf16(false, pf.v, false, vf,
                                                      (short)0, o[dt], false, false);
    }
  }

  // normalize and write concat-head layout Ob[B,S,H*D] (bf16)
#pragma unroll
  for (int j = 0; j < 8; ++j) {
    const float inv = 1.0f / l[j];
    const int row = q0 + hf * 8 + j;
    bf16_t* dst = Ob + ((size_t)b * S + row) * EO + h * D;
#pragma unroll
    for (int dt = 0; dt < 16; ++dt)
      dst[dt * 16 + lm] = f2bf(o[dt][j] * inv);
  }
}

// ---------------------------------------------------------------------------
// Kernel 3: Out[B*S,768] = Ob @ Wo + bo (fp32 out).
// grid: (B*S/16, 48/4), block 128 (4 waves, one 16x16 tile per wave).
// ---------------------------------------------------------------------------
__global__ __launch_bounds__(128)
void out_proj_kernel(const bf16_t* __restrict__ Ob, const bf16_t* __restrict__ WoT,
                     const float* __restrict__ bo, float* __restrict__ Out)
{
  const int wave = threadIdx.x >> 5, lane = threadIdx.x & 31;
  const int hf = lane >> 4, lm = lane & 15;
  const int r0 = blockIdx.x * 16;
  const int n0 = (blockIdx.y * 4 + wave) * 16;

  const bf16_t* arow = Ob  + (size_t)(r0 + lm) * EO;
  const bf16_t* brow = WoT + (size_t)(n0 + lm) * EO;

  v8f acc = {};
  for (int kc = 0; kc < EO; kc += 32) {
    V16 a;
    a.h[0] = *(const v8bf*)(arow + kc + hf * 16);
    a.h[1] = *(const v8bf*)(arow + kc + hf * 16 + 8);
    v16bf bfr = *(const v16bf*)(brow + kc + hf * 16);
    acc = __builtin_amdgcn_wmma_f32_16x16x32_bf16(false, a.v, false, bfr,
                                                  (short)0, acc, false, false);
  }
  const float bias = bo[n0 + lm];
#pragma unroll
  for (int j = 0; j < 8; ++j) {
    const int row = r0 + hf * 8 + j;
    Out[(size_t)row * EO + n0 + lm] = acc[j] + bias;
  }
}

// ---------------------------------------------------------------------------
extern "C" void kernel_launch(void* const* d_in, const int* in_sizes, int n_in,
                              void* d_out, int out_size, void* d_ws, size_t ws_size,
                              hipStream_t stream)
{
  (void)in_sizes; (void)n_in; (void)out_size; (void)ws_size;
  const float* Xk  = (const float*)d_in[0];
  const float* Xv  = (const float*)d_in[1];
  const float* Xq  = (const float*)d_in[2];
  const float* MKp = (const float*)d_in[3];
  const float* MVp = (const float*)d_in[4];
  const float* MQp = (const float*)d_in[5];
  const float* Wo  = (const float*)d_in[6];
  const float* bo  = (const float*)d_in[7];
  float* Out = (float*)d_out;

  char* ws = (char*)d_ws;
  const size_t mt_bytes  = (size_t)3 * H * D * E * sizeof(bf16_t);  //  3.54 MB
  const size_t wot_bytes = (size_t)EO * EO * sizeof(bf16_t);        //  1.18 MB
  const size_t qkv_bytes = (size_t)B * H * S * D * sizeof(bf16_t);  // 12.58 MB
  bf16_t* Mt  = (bf16_t*)(ws);
  bf16_t* WoT = (bf16_t*)(ws + mt_bytes);
  bf16_t* Qb  = (bf16_t*)(ws + mt_bytes + wot_bytes);
  bf16_t* Kb  = (bf16_t*)(ws + mt_bytes + wot_bytes + qkv_bytes);
  bf16_t* Vt  = (bf16_t*)(ws + mt_bytes + wot_bytes + 2 * qkv_bytes);
  bf16_t* Ob  = (bf16_t*)(ws + mt_bytes + wot_bytes + 3 * qkv_bytes); // ~55 MB total

  dim3 g0((H * D * E + 255) / 256, 1, 4);
  convert_weights_kernel<<<g0, 256, 0, stream>>>(MKp, MVp, MQp, Wo, Mt, WoT);

  dim3 g1(S / 32, B * H, 3);
  qkv_proj_kernel<<<g1, 256, 0, stream>>>(Xk, Xv, Xq, Mt, Kb, Vt, Qb);

  dim3 g2(S / 64, B * H);
  attn_kernel<<<g2, 128, 0, stream>>>(Qb, Kb, Vt, Ob);

  dim3 g4((B * S) / 16, (EO / 16) / 4);
  out_proj_kernel<<<g4, 128, 0, stream>>>(Ob, WoT, bo, Out);
}
